// SA_Conv2d_12661563588823
// MI455X (gfx1250) — matrix operand order repaired and checked by reading, not compiled, not run on hardware
//
#include <hip/hip_runtime.h>
#include <math.h>

typedef __attribute__((ext_vector_type(16))) __bf16 v16bf;
typedef __attribute__((ext_vector_type(8)))  float  v8f;

union Frag {
    v16bf v;
    uint4 q[2];
    unsigned int w[8];
    unsigned short u[16];
};

__device__ __forceinline__ unsigned short f2bf(float f) {
    union { float f; unsigned int u; } c; c.f = f;
    unsigned int r = c.u + 0x7FFFu + ((c.u >> 16) & 1u);
    return (unsigned short)(r >> 16);
}

// K index inside a 16x32 bf16 A-fragment for (vgpr v, half, lane-group g2)
__device__ __forceinline__ int a_k(int v, int half, int g2) {
    return 2 * v + half + 16 * g2;
}

__device__ __forceinline__ v8f vzero() {
    v8f z;
#pragma unroll
    for (int i = 0; i < 8; ++i) z[i] = 0.0f;
    return z;
}

__device__ __forceinline__ v8f wmma_bf16(v16bf a, v16bf b, v8f c) {
    return __builtin_amdgcn_wmma_f32_16x16x32_bf16(false, a, false, b, (short)0, c, false, false);
}

// ---------------------------------------------------------------------------
// Constants
#define B_  16
#define C_  256
#define HW_ 4096
#define HWP_ 1024
#define CK_ 32
#define CH_ 128
#define OC_ 192   // 32 f + 32 g + 128 h stacked

// ---------------------------------------------------------------------------
// K0: tiled transpose x (B,C,HW) f32 -> xt (B,HW,C) bf16
__global__ void __launch_bounds__(256) k_transpose(const float* __restrict__ x,
                                                   unsigned short* __restrict__ xt) {
    __shared__ float tile[32][33];
    int blk = blockIdx.x;             // b * 8 * 128 blocks
    int hwt = blk % 128;
    int ct  = (blk / 128) % 8;
    int b   = blk / (128 * 8);
#pragma unroll
    for (int i = 0; i < 4; ++i) {
        int lin = threadIdx.x + 256 * i;
        int ci = lin >> 5, hi = lin & 31;
        tile[ci][hi] = x[((size_t)(b * C_) + ct * 32 + ci) * HW_ + hwt * 32 + hi];
    }
    __syncthreads();
#pragma unroll
    for (int i = 0; i < 4; ++i) {
        int lin = threadIdx.x + 256 * i;
        int ho = lin >> 5, co = lin & 31;
        xt[((size_t)(b * HW_) + hwt * 32 + ho) * C_ + ct * 32 + co] = f2bf(tile[co][ho]);
    }
}

// ---------------------------------------------------------------------------
// K1: pack weights into A-fragment layout + stack biases
// wall: 12 rowtiles x 8 kchunks x 512 ; wvp: 16 rowtiles x 4 kchunks x 512
__global__ void __launch_bounds__(256) k_pack_w(const float* __restrict__ wf,
                                                const float* __restrict__ wg,
                                                const float* __restrict__ wh,
                                                const float* __restrict__ wvm,
                                                const float* __restrict__ bf,
                                                const float* __restrict__ bg,
                                                const float* __restrict__ bh,
                                                unsigned short* __restrict__ wall,
                                                unsigned short* __restrict__ wvp,
                                                float* __restrict__ biasAll) {
    int idx = blockIdx.x * 256 + threadIdx.x;
    const int NW = 12 * 8 * 512;          // 49152
    const int NV = 16 * 4 * 512;          // 32768
    if (idx < NW) {
        int tile = idx / 512, within = idx % 512;
        int lane = within / 16, j = within % 16;
        int m = lane & 15, g2 = lane >> 4;
        int k = a_k(j >> 1, j & 1, g2);
        int rt = tile / 8, kc = tile % 8;
        int r = rt * 16 + m, c = kc * 32 + k;
        float val = (r < 32) ? wf[r * C_ + c]
                  : (r < 64) ? wg[(r - 32) * C_ + c]
                             : wh[(r - 64) * C_ + c];
        wall[idx] = f2bf(val);
    } else if (idx < NW + NV) {
        int i2 = idx - NW;
        int tile = i2 / 512, within = i2 % 512;
        int lane = within / 16, j = within % 16;
        int m = lane & 15, g2 = lane >> 4;
        int k = a_k(j >> 1, j & 1, g2);
        int rt = tile / 4, kc = tile % 4;
        int r = rt * 16 + m, c = kc * 32 + k;
        wvp[i2] = f2bf(wvm[r * CH_ + c]);
    } else if (idx < NW + NV + OC_) {
        int i3 = idx - NW - NV;
        biasAll[i3] = (i3 < 32) ? bf[i3] : (i3 < 64) ? bg[i3 - 32] : bh[i3 - 64];
    }
}

// ---------------------------------------------------------------------------
// K2: fused f/g/h 1x1 conv:  Y(b, 0..191, hw) = Wall x[b] + bias   (WMMA)
// one wave: 16 rows x 64 cols, K=256
__global__ void __launch_bounds__(256) k_conv_fgh(const unsigned short* __restrict__ xt,
                                                  const unsigned short* __restrict__ wall,
                                                  const float* __restrict__ biasAll,
                                                  float* __restrict__ Y) {
    int lane = threadIdx.x & 31;
    int w = (blockIdx.x * 256 + threadIdx.x) >> 5;   // 0..12287
    int n4 = w & 63;
    int rt = (w >> 6) % 12;
    int b  = w / (64 * 12);
    int nlane = lane & 15, g2 = lane >> 4;

    v8f acc[4];
#pragma unroll
    for (int nt = 0; nt < 4; ++nt) acc[nt] = vzero();

#pragma unroll
    for (int kc = 0; kc < 8; ++kc) {
        Frag a;
        const uint4* ap = (const uint4*)(wall + ((size_t)(rt * 8 + kc) * 512 + lane * 16));
        a.q[0] = ap[0]; a.q[1] = ap[1];
#pragma unroll
        for (int nt = 0; nt < 4; ++nt) {
            int hw = n4 * 64 + nt * 16 + nlane;
            Frag bx;
            const uint4* bp = (const uint4*)(xt + ((size_t)(b * HW_) + hw) * C_ + kc * 32 + g2 * 16);
            bx.q[0] = bp[0]; bx.q[1] = bp[1];
            acc[nt] = wmma_bf16(a.v, bx.v, acc[nt]);
        }
    }
    float bias_v[8];
#pragma unroll
    for (int v = 0; v < 8; ++v) bias_v[v] = biasAll[rt * 16 + v + 8 * g2];
#pragma unroll
    for (int nt = 0; nt < 4; ++nt) {
        int hw = n4 * 64 + nt * 16 + nlane;
#pragma unroll
        for (int v = 0; v < 8; ++v) {
            int oc = rt * 16 + v + 8 * g2;
            Y[((size_t)(b * OC_) + oc) * HW_ + hw] = acc[nt][v] + bias_v[v];
        }
    }
}

// ---------------------------------------------------------------------------
// K3a: pack f into A fragments (per 16-query block, pre-swizzled)
__global__ void __launch_bounds__(256) k_pack_f(const float* __restrict__ Y,
                                                unsigned short* __restrict__ fpack) {
    int idx = blockIdx.x * 256 + threadIdx.x;     // 16*256*512
    int j = idx & 15;
    int lane = (idx >> 4) & 31;
    int blk = idx >> 9;
    int qb = blk & 255, b = blk >> 8;
    int m = lane & 15, g2 = lane >> 4;
    int c = a_k(j >> 1, j & 1, g2);               // channel 0..31
    int q = qb * 16 + m;
    fpack[idx] = f2bf(Y[((size_t)(b * OC_) + c) * HW_ + q]);
}

// K3b: maxpool g -> gt (B,1024,32) bf16 (key-major)
__global__ void __launch_bounds__(256) k_pool_g(const float* __restrict__ Y,
                                                unsigned short* __restrict__ gt) {
    int idx = blockIdx.x * 256 + threadIdx.x;     // 16*1024*32
    int c = idx & 31;
    int key = (idx >> 5) & 1023;
    int b = idx >> 15;
    int hp = key >> 5, wp = key & 31;
    size_t base = ((size_t)(b * OC_) + 32 + c) * HW_;
    int hw = hp * 128 + wp * 2;
    float v0 = Y[base + hw],      v1 = Y[base + hw + 1];
    float v2 = Y[base + hw + 64], v3 = Y[base + hw + 65];
    gt[idx] = f2bf(fmaxf(fmaxf(v0, v1), fmaxf(v2, v3)));
}

// K3c: maxpool h -> ht (B,128,1024) bf16 (channel-major)
__global__ void __launch_bounds__(256) k_pool_h(const float* __restrict__ Y,
                                                unsigned short* __restrict__ ht) {
    int idx = blockIdx.x * 256 + threadIdx.x;     // 16*128*1024
    int key = idx & 1023;
    int c = (idx >> 10) & 127;
    int b = idx >> 17;
    int hp = key >> 5, wp = key & 31;
    size_t base = ((size_t)(b * OC_) + 64 + c) * HW_;
    int hw = hp * 128 + wp * 2;
    float v0 = Y[base + hw],      v1 = Y[base + hw + 1];
    float v2 = Y[base + hw + 64], v3 = Y[base + hw + 65];
    ht[idx] = f2bf(fmaxf(fmaxf(v0, v1), fmaxf(v2, v3)));
}

// ---------------------------------------------------------------------------
// K4: flash attention. One wave = 16 queries. 32-key chunks, online softmax.
// s = f^T g (K=32 in one WMMA), o^T += P x h^T (8 channel tiles).
__global__ void __launch_bounds__(256) k_attn(const unsigned short* __restrict__ fpack,
                                              const unsigned short* __restrict__ gt,
                                              const unsigned short* __restrict__ ht,
                                              unsigned short* __restrict__ ot) {
    __shared__ unsigned short plds[8][512];       // per-wave 16x32 bf16 P tile
    int lane = threadIdx.x & 31;
    int wid = threadIdx.x >> 5;
    int w = (blockIdx.x * 256 + threadIdx.x) >> 5;  // 0..4095
    int qb = w & 255, b = w >> 8;
    int nlane = lane & 15, g2 = lane >> 4;

    Frag af;
    {
        const uint4* fp = (const uint4*)(fpack + ((size_t)(b * 256 + qb) * 512 + lane * 16));
        af.q[0] = fp[0]; af.q[1] = fp[1];
    }

    v8f oacc[8];
#pragma unroll
    for (int t = 0; t < 8; ++t) oacc[t] = vzero();
    float mrow[8], lrow[8];
#pragma unroll
    for (int v = 0; v < 8; ++v) { mrow[v] = -INFINITY; lrow[v] = 0.0f; }

    for (int kc = 0; kc < 32; ++kc) {
        int k0 = kc * 32;
        // logits: two 16x16 tiles, K = 32 channels in one WMMA each
        Frag bg1, bg2;
        {
            const uint4* gp1 = (const uint4*)(gt + ((size_t)(b * HWP_) + k0 + nlane) * CK_ + g2 * 16);
            bg1.q[0] = gp1[0]; bg1.q[1] = gp1[1];
            const uint4* gp2 = (const uint4*)(gt + ((size_t)(b * HWP_) + k0 + 16 + nlane) * CK_ + g2 * 16);
            bg2.q[0] = gp2[0]; bg2.q[1] = gp2[1];
        }
        v8f s1 = wmma_bf16(af.v, bg1.v, vzero());
        v8f s2 = wmma_bf16(af.v, bg2.v, vzero());

        // online softmax per row (rows v+8*g2 live in this half-wave)
#pragma unroll
        for (int v = 0; v < 8; ++v) {
            float t = fmaxf(s1[v], s2[v]);
#pragma unroll
            for (int off = 1; off < 16; off <<= 1) t = fmaxf(t, __shfl_xor(t, off, 16));
            float mnew = fmaxf(mrow[v], t);
            float sc = __expf(mrow[v] - mnew);
            float p1 = __expf(s1[v] - mnew);
            float p2 = __expf(s2[v] - mnew);
            float rs = p1 + p2;
#pragma unroll
            for (int off = 1; off < 16; off <<= 1) rs += __shfl_xor(rs, off, 16);
            lrow[v] = lrow[v] * sc + rs;
            mrow[v] = mnew;
#pragma unroll
            for (int t8 = 0; t8 < 8; ++t8) oacc[t8][v] *= sc;
            int m = v + 8 * g2;
            plds[wid][m * 32 + nlane]      = f2bf(p1);
            plds[wid][m * 32 + 16 + nlane] = f2bf(p2);
        }

        // D-layout -> A-layout transpose via per-wave LDS slice
        Frag ap;
        int mq = lane & 15;
#pragma unroll
        for (int v = 0; v < 8; ++v) {
            int kb = a_k(v, 0, g2);
            ap.w[v] = *(const unsigned int*)&plds[wid][mq * 32 + kb];
        }

        // o^T accumulate: 8 channel tiles of h^T (32 keys x 16 channels)
#pragma unroll
        for (int t = 0; t < 8; ++t) {
            int ch = t * 16 + nlane;
            Frag bh;
            const uint4* hp = (const uint4*)(ht + ((size_t)(b * CH_) + ch) * HWP_ + k0 + g2 * 16);
            bh.q[0] = hp[0]; bh.q[1] = hp[1];
            oacc[t] = wmma_bf16(ap.v, bh.v, oacc[t]);
        }
    }

    float inv[8];
#pragma unroll
    for (int v = 0; v < 8; ++v) inv[v] = 1.0f / lrow[v];
#pragma unroll
    for (int t = 0; t < 8; ++t)
#pragma unroll
        for (int v = 0; v < 8; ++v) {
            int m = v + 8 * g2;
            int hw = qb * 16 + m;
            ot[((size_t)(b * HW_) + hw) * CH_ + t * 16 + nlane] = f2bf(oacc[t][v] * inv[v]);
        }
}

// ---------------------------------------------------------------------------
// K5: out = gamma * (wv . o) + bv + x   (WMMA, fused epilogue)
__global__ void __launch_bounds__(256) k_conv_out(const unsigned short* __restrict__ wvp,
                                                  const unsigned short* __restrict__ ot,
                                                  const float* __restrict__ bv,
                                                  const float* __restrict__ x,
                                                  const float* __restrict__ gamma,
                                                  float* __restrict__ out) {
    int lane = threadIdx.x & 31;
    int w = (blockIdx.x * 256 + threadIdx.x) >> 5;   // 16*16*64 waves
    int n4 = w & 63;
    int rt = (w >> 6) & 15;
    int b  = w >> 10;
    int nlane = lane & 15, g2 = lane >> 4;

    v8f acc[4];
#pragma unroll
    for (int nt = 0; nt < 4; ++nt) acc[nt] = vzero();

#pragma unroll
    for (int kc = 0; kc < 4; ++kc) {
        Frag a;
        const uint4* ap = (const uint4*)(wvp + ((size_t)(rt * 4 + kc) * 512 + lane * 16));
        a.q[0] = ap[0]; a.q[1] = ap[1];
#pragma unroll
        for (int nt = 0; nt < 4; ++nt) {
            int hw = n4 * 64 + nt * 16 + nlane;
            Frag bo;
            const uint4* bp = (const uint4*)(ot + ((size_t)(b * HW_) + hw) * CH_ + kc * 32 + g2 * 16);
            bo.q[0] = bp[0]; bo.q[1] = bp[1];
            acc[nt] = wmma_bf16(a.v, bo.v, acc[nt]);
        }
    }
    float gm = gamma[0];
    float bias_v[8];
#pragma unroll
    for (int v = 0; v < 8; ++v) bias_v[v] = bv[rt * 16 + v + 8 * g2];
#pragma unroll
    for (int nt = 0; nt < 4; ++nt) {
        int hw = n4 * 64 + nt * 16 + nlane;
#pragma unroll
        for (int v = 0; v < 8; ++v) {
            int oc = rt * 16 + v + 8 * g2;
            size_t idx = ((size_t)(b * C_) + oc) * HW_ + hw;
            out[idx] = gm * acc[nt][v] + bias_v[v] + x[idx];
        }
    }
}

// ---------------------------------------------------------------------------
extern "C" void kernel_launch(void* const* d_in, const int* in_sizes, int n_in,
                              void* d_out, int out_size, void* d_ws, size_t ws_size,
                              hipStream_t stream) {
    const float* x   = (const float*)d_in[0];
    const float* wf  = (const float*)d_in[1];
    const float* bf  = (const float*)d_in[2];
    const float* wg  = (const float*)d_in[3];
    const float* bg  = (const float*)d_in[4];
    const float* wh  = (const float*)d_in[5];
    const float* bh  = (const float*)d_in[6];
    const float* wvm = (const float*)d_in[7];
    const float* bv  = (const float*)d_in[8];
    const float* gm  = (const float*)d_in[9];
    float* out = (float*)d_out;

    char* ws = (char*)d_ws;
    size_t off = 0;
    unsigned short* xt    = (unsigned short*)(ws + off); off += (size_t)B_ * HW_ * C_ * 2;   // 33.5 MB
    unsigned short* wall  = (unsigned short*)(ws + off); off += 12 * 8 * 512 * 2;            // 96 KB
    unsigned short* wvp   = (unsigned short*)(ws + off); off += 16 * 4 * 512 * 2;            // 64 KB
    float*          biasA = (float*)(ws + off);          off += 1024;
    float*          Y     = (float*)(ws + off);          off += (size_t)B_ * OC_ * HW_ * 4;  // 50 MB
    unsigned short* fpack = (unsigned short*)(ws + off); off += (size_t)B_ * 256 * 512 * 2;  // 4 MB
    unsigned short* gt    = (unsigned short*)(ws + off); off += (size_t)B_ * HWP_ * CK_ * 2; // 1 MB
    unsigned short* ht    = (unsigned short*)(ws + off); off += (size_t)B_ * CH_ * HWP_ * 2; // 4 MB
    unsigned short* ot    = (unsigned short*)(ws + off); off += (size_t)B_ * HW_ * CH_ * 2;  // 16 MB

    // K0: transpose/convert x
    k_transpose<<<16 * 8 * 128, 256, 0, stream>>>(x, xt);
    // K1: pack weights + biases
    k_pack_w<<<(12 * 8 * 512 + 16 * 4 * 512 + 256 + 255) / 256, 256, 0, stream>>>(
        wf, wg, wh, wvm, bf, bg, bh, wall, wvp, biasA);
    // K2: fused f/g/h conv (WMMA)
    k_conv_fgh<<<(16 * 12 * 64 * 32) / 256, 256, 0, stream>>>(xt, wall, biasA, Y);
    // K3: packing / pooling
    k_pack_f<<<(16 * 256 * 512) / 256, 256, 0, stream>>>(Y, fpack);
    k_pool_g<<<(16 * 1024 * 32) / 256, 256, 0, stream>>>(Y, gt);
    k_pool_h<<<(16 * 128 * 1024) / 256, 256, 0, stream>>>(Y, ht);
    // K4: flash attention (WMMA + online softmax)
    k_attn<<<(16 * 256 * 32) / 256, 256, 0, stream>>>(fpack, gt, ht, ot);
    // K5: output projection + residual (WMMA)
    k_conv_out<<<(16 * 16 * 64 * 32) / 256, 256, 0, stream>>>(wvp, ot, bv, x, gm, out);
}
